// RelativeMultiHeadAttn_28776280883722
// MI455X (gfx1250) — matrix operand order repaired and checked by reading, not compiled, not run on hardware
//
#include <hip/hip_runtime.h>

// ---------------------------------------------------------------------------
// Relative multi-head attention (Transformer-XL style) for MI455X / gfx1250.
//
// Pipeline (all on `stream`, workspace assumed >= ~30 MB):
//   1) pos_table_kernel : sinusoidal relative-position table [2048,64] -> bf16
//   2) cvt_x_kernel     : x  fp32 -> bf16            [4096,1024]
//   3) cvt_wT_kernel    : Wqv fp32 -> bf16 TRANSPOSED [2048,1024]  (n-major)
//   4) qv_gemm_kernel   : qv = x @ Wqv via v_wmma_f32_16x16x32_bf16
//   5) rel_attn_kernel  : flash-style fused attention with online softmax.
//      NEW: the 4 waves of a block share (b,h); K/V 32-key chunks are staged
//      into LDS once per block with double-buffered GLOBAL_LOAD_ASYNC_TO_LDS
//      (ASYNCcnt + barrier), overlapping global traffic with WMMA compute and
//      cutting K/V HBM traffic 4x.
// ---------------------------------------------------------------------------

#define B_  4
#define L_  1024
#define D_  1024
#define H_  16
#define HD_ 64

typedef __attribute__((ext_vector_type(16))) __bf16 v16bf;
typedef __attribute__((ext_vector_type(8)))  __bf16 v8bf;
typedef __attribute__((ext_vector_type(8)))  float  v8f;

static __device__ __forceinline__ unsigned short f2bfbits(float f) {
  unsigned u = __builtin_bit_cast(unsigned, f);
  return (unsigned short)((u + 0x7FFFu + ((u >> 16) & 1u)) >> 16); // RNE
}
static __device__ __forceinline__ __bf16 f2bf(float f) {
  unsigned short s = f2bfbits(f);
  return __builtin_bit_cast(__bf16, s);
}
static __device__ __forceinline__ float bf2f(__bf16 b) {
  unsigned short s = __builtin_bit_cast(unsigned short, b);
  unsigned u = ((unsigned)s) << 16;
  return __builtin_bit_cast(float, u);
}
static __device__ __forceinline__ v8f wmma_bf16(v16bf a, v16bf b, v8f c) {
  // D(16x16 f32) = A(16x32 bf16) * B(32x16 bf16) + C
  return __builtin_amdgcn_wmma_f32_16x16x32_bf16(
      /*neg_a=*/false, a, /*neg_b=*/false, b,
      /*c_mod=*/(short)0, c, /*reuse_a=*/false, /*reuse_b=*/false);
}
static __device__ __forceinline__ unsigned long long pack4(float a, float b,
                                                           float c, float d) {
  return (unsigned long long)f2bfbits(a)
       | ((unsigned long long)f2bfbits(b) << 16)
       | ((unsigned long long)f2bfbits(c) << 32)
       | ((unsigned long long)f2bfbits(d) << 48);
}

// Async 16B copy global -> LDS (ASYNCcnt-tracked). LDS byte address is the low
// 32 bits of the generic pointer (flat->LDS mapping, ISA 10.2).
static __device__ __forceinline__ void async_b128(void* lds, const void* g) {
  unsigned ldsa = (unsigned)(size_t)lds;
  asm volatile("global_load_async_to_lds_b128 %0, %1, off"
               :: "v"(ldsa), "v"(g) : "memory");
}
static __device__ __forceinline__ void wait_async0() {
  asm volatile("s_wait_asynccnt 0x0" ::: "memory");
}

// ---------------------------------------------------------------------------
// 1) relative sinusoidal position table: pos[l,d], l in [0,2048), d in [0,64)
__global__ void pos_table_kernel(__bf16* __restrict__ pos) {
  int id = blockIdx.x * blockDim.x + threadIdx.x;
  if (id >= 2 * L_ * HD_) return;
  int l = id >> 6, d = id & 63;
  float val = 0.0f;
  if (l != 0) {
    float p = (float)(l - L_);
    int i = d & 31;
    float freq = expf(-(float)i * (logf(10000.0f) / 31.0f));
    float a = p * freq;
    val = (d < 32) ? sinf(a) : cosf(a);
  }
  pos[id] = f2bf(val);
}

// 2) x fp32 -> bf16
__global__ void cvt_x_kernel(const float* __restrict__ x,
                             __bf16* __restrict__ xb) {
  int id = blockIdx.x * blockDim.x + threadIdx.x;
  if (id < B_ * L_ * D_) xb[id] = f2bf(x[id]);
}

// 3) Wqv fp32 [1024,2048] -> bf16 transposed wT [2048,1024]
__global__ void cvt_wT_kernel(const float* __restrict__ W,
                              __bf16* __restrict__ wT) {
  int id = blockIdx.x * blockDim.x + threadIdx.x;
  if (id >= 2 * D_ * D_) return;
  int n = id >> 10, k = id & 1023;
  wT[id] = f2bf(W[(size_t)k * (2 * D_) + n]);
}

// ---------------------------------------------------------------------------
// 4) qv GEMM: [4096,1024] x [1024,2048]; one 16x16 tile per wave, K in 32-steps.
__global__ __launch_bounds__(128) void qv_gemm_kernel(
    const __bf16* __restrict__ xb, const __bf16* __restrict__ wT,
    __bf16* __restrict__ qb, __bf16* __restrict__ vT) {
  const int lane = threadIdx.x & 31;
  const int wave = threadIdx.x >> 5;
  const int hi   = lane >> 4;
  const int l15  = lane & 15;
  const int n0   = blockIdx.x * 16;
  const int m0   = (blockIdx.y * 4 + wave) * 16;

  v8f acc = {0.f, 0.f, 0.f, 0.f, 0.f, 0.f, 0.f, 0.f};
  const __bf16* arow = xb + (size_t)(m0 + l15) * D_;                 // A: lane = row
  const __bf16* bcol = wT + (size_t)(n0 + l15) * D_ + (hi ? 16 : 0); // B: lane = col

  for (int kk = 0; kk < D_; kk += 32) {
    // A 16x32 bf16: lane<16 -> K 0..15, lane>=16 -> K 16..31 (contiguous)
    v8bf a_lo = *(const v8bf*)(arow + kk + (hi ? 16 : 0));
    v8bf a_hi = *(const v8bf*)(arow + kk + (hi ? 24 : 8));
    v16bf a;
#pragma unroll
    for (int i = 0; i < 8; ++i) { a[i] = a_lo[i]; a[i + 8] = a_hi[i]; }
    // B 32x16 bf16: lane<16 -> K 0..15, lane>=16 -> K 16..31 (contiguous)
    v16bf b = *(const v16bf*)(bcol + kk);
    acc = wmma_bf16(a, b, acc);
  }

  // D layout: element (vgpr r, lane) = (row m0 + r + 8*hi, col n0 + l15)
  if (n0 < D_) {                      // q half -> qb row-major bf16
#pragma unroll
    for (int r = 0; r < 8; ++r)
      qb[(size_t)(m0 + r + 8 * hi) * D_ + n0 + l15] = f2bf(acc[r]);
  } else {                            // v half -> vT[b][h*64+d][l]
    int bidx = m0 >> 10, lbase = m0 & 1023;
    int ch = n0 - D_ + l15;           // h*64+d
#pragma unroll
    for (int r = 0; r < 8; ++r)
      vT[((size_t)bidx * D_ + ch) * L_ + lbase + r + 8 * hi] = f2bf(acc[r]);
  }
}

// ---------------------------------------------------------------------------
// 5) fused attention. One wave = 16 queries of one (b,h); block = 4 waves.
//
// Per 32-key chunk the block shares K [32 j][64 d] (4 KB) and V [64 ch][32 l]
// (4 KB) in LDS, filled by double-buffered async loads (4 x b128 per thread).
static __device__ __forceinline__ void issue_chunk(
    const __bf16* kbase, const __bf16* vbh, int j0,
    __bf16* kdst, __bf16* vdst, int tid) {
#pragma unroll
  for (int rep = 0; rep < 2; ++rep) {
    int u = tid + rep * 128;                    // 16B units, 256 each
    // K: row j = u>>3 (32 rows of 128B)
    async_b128(kdst + u * 8,
               (const unsigned char*)(kbase + (size_t)(j0 + (u >> 3)) * D_) +
                   (u & 7) * 16);
    // V: row ch = u>>2 (64 rows of 64B)
    async_b128(vdst + u * 8,
               (const unsigned char*)(vbh + (size_t)(u >> 2) * L_ + j0) +
                   (u & 3) * 16);
  }
}

// Score subtile (16q x 16j): AC (2 wmma) + shifted BD via band-GEMM
// G = Aw * pos[c-15+t] (4 wmma) + shuffle gather  BD[m,n] = G[m, n-m+15].
static __device__ __forceinline__ void score_subtile(
    const v16bf* Aq, const v16bf* Aw, const __bf16* ldsK,
    const __bf16* pos, const int* maskrow, int jt, int jtloc, int q0, int lane,
    float s[8]) {
  const int hi = lane >> 4, l15 = lane & 15;
  v8f acc = {0.f, 0.f, 0.f, 0.f, 0.f, 0.f, 0.f, 0.f};
  const __bf16* krow = ldsK + (size_t)(jtloc + l15) * HD_;
#pragma unroll
  for (int c = 0; c < 2; ++c) {
    v16bf bk = *(const v16bf*)(krow + c * 32 + (hi ? 16 : 0));
    acc = wmma_bf16(Aq[c], bk, acc);
  }
  const int csh = L_ + jt - q0;                // rel = csh + n - m
  int rel0 = csh - 15 + l15;                   // G0 columns t = 0..15
  int rel1 = csh + 1 + l15;                    // G1 columns t = 16..31
  rel0 = rel0 < 0 ? 0 : (rel0 > 2 * L_ - 1 ? 2 * L_ - 1 : rel0);
  rel1 = rel1 < 0 ? 0 : (rel1 > 2 * L_ - 1 ? 2 * L_ - 1 : rel1);
  __builtin_prefetch(pos + (size_t)(rel1 + 16) * HD_, 0, 1); // next pos band
  v8f g0 = {0.f, 0.f, 0.f, 0.f, 0.f, 0.f, 0.f, 0.f};
  v8f g1 = {0.f, 0.f, 0.f, 0.f, 0.f, 0.f, 0.f, 0.f};
#pragma unroll
  for (int c = 0; c < 2; ++c) {
    v16bf bp0 = *(const v16bf*)(pos + (size_t)rel0 * HD_ + c * 32 + (hi ? 16 : 0));
    g0 = wmma_bf16(Aw[c], bp0, g0);
    v16bf bp1 = *(const v16bf*)(pos + (size_t)rel1 * HD_ + c * 32 + (hi ? 16 : 0));
    g1 = wmma_bf16(Aw[c], bp1, g1);
  }
  const float madd = (maskrow[jt + l15] == 0) ? -1e30f : 0.0f;
#pragma unroll
  for (int r = 0; r < 8; ++r) {
    int t  = l15 - r - 8 * hi + 15;            // n - m + 15, in [0,30]
    int sl = (t & 15) + (hi ? 16 : 0);         // same vgpr row, shifted lane
    float v0 = __shfl(g0[r], sl, 32);
    float v1 = __shfl(g1[r], sl, 32);
    s[r] = acc[r] + ((t < 16) ? v0 : v1) + madd;
  }
}

__global__ __launch_bounds__(128) void rel_attn_kernel(
    const __bf16* __restrict__ qb, const __bf16* __restrict__ xb,
    const __bf16* __restrict__ vT, const __bf16* __restrict__ pos,
    const float* __restrict__ rr, const float* __restrict__ rw,
    const int* __restrict__ mask, float* __restrict__ out) {
  __shared__ __align__(16) __bf16 ldsK[2][32 * HD_];  // [buf][j][d]   4 KB each
  __shared__ __align__(16) __bf16 ldsV[2][HD_ * 32];  // [buf][ch][l]  4 KB each
  __shared__ unsigned short smemP[4][512];            // wave-private P stage

  const int tid  = threadIdx.x;
  const int lane = tid & 31;
  const int wave = tid >> 5;
  const int hi = lane >> 4, l15 = lane & 15;
  const int bh = blockIdx.x, b = bh >> 4, h = bh & 15;
  const int q0 = blockIdx.y * 64 + wave * 16;

  // Q tile -> two A operand sets (AC bias = r_r_bias, BD bias = r_w_bias)
  v16bf Aq[2], Aw[2];
  const __bf16* qrow = qb + (size_t)(b * L_ + q0 + l15) * D_ + h * HD_;
#pragma unroll
  for (int c = 0; c < 2; ++c) {
    int blo = c * 32 + (hi ? 16 : 0);
    int bhi = c * 32 + (hi ? 24 : 8);
#pragma unroll
    for (int i = 0; i < 8; ++i) {
      float qlo = bf2f(qrow[blo + i]);
      float qhi = bf2f(qrow[bhi + i]);
      Aq[c][i]     = f2bf(qlo + rr[h * HD_ + blo + i]);
      Aq[c][i + 8] = f2bf(qhi + rr[h * HD_ + bhi + i]);
      Aw[c][i]     = f2bf(qlo + rw[h * HD_ + blo + i]);
      Aw[c][i + 8] = f2bf(qhi + rw[h * HD_ + bhi + i]);
    }
  }

  v8f o[4];
#pragma unroll
  for (int s4 = 0; s4 < 4; ++s4)
#pragma unroll
    for (int r = 0; r < 8; ++r) o[s4][r] = 0.0f;
  float mrow[8], lrow[8];
#pragma unroll
  for (int r = 0; r < 8; ++r) { mrow[r] = -3.0e38f; lrow[r] = 0.0f; }

  const __bf16* kbase   = xb + (size_t)(b * L_) * D_ + h * HD_;
  const __bf16* vbh     = vT + ((size_t)b * D_ + h * HD_) * L_;
  const int*    maskrow = mask + b * L_;
  unsigned short* sm = &smemP[wave][0];

  issue_chunk(kbase, vbh, 0, &ldsK[0][0], &ldsV[0][0], tid); // prologue

  for (int j0 = 0; j0 < L_; j0 += 32) {
    const int buf = (j0 >> 5) & 1;
    wait_async0();        // this wave's chunk landed
    __syncthreads();      // all waves' chunk landed; prior reads of buf done
    if (j0 + 32 < L_)     // overlap next chunk's DMA with this chunk's WMMAs
      issue_chunk(kbase, vbh, j0 + 32, &ldsK[buf ^ 1][0], &ldsV[buf ^ 1][0], tid);

    float sA[8], sB[8];
    score_subtile(Aq, Aw, &ldsK[buf][0], pos, maskrow, j0,      0,  q0, lane, sA);
    score_subtile(Aq, Aw, &ldsK[buf][0], pos, maskrow, j0 + 16, 16, q0, lane, sB);

    // online softmax update over the 32-key chunk
#pragma unroll
    for (int r = 0; r < 8; ++r) {
      float rm = fmaxf(sA[r], sB[r]);
#pragma unroll
      for (int off = 1; off < 16; off <<= 1) rm = fmaxf(rm, __shfl_xor(rm, off, 32));
      float mnew = fmaxf(mrow[r], rm);
      float sc = __expf(mrow[r] - mnew);
      mrow[r] = mnew;
      float pA = __expf(sA[r] - mnew);
      float pB = __expf(sB[r] - mnew);
      lrow[r] = lrow[r] * sc + pA + pB;
#pragma unroll
      for (int s4 = 0; s4 < 4; ++s4) o[s4][r] *= sc;
      sA[r] = pA; sB[r] = pB;
    }

    // P (D-layout) -> LDS [jl][q] -> reload as A-operand (transpose via LDS)
    {
      unsigned long long* d0 = (unsigned long long*)(sm + (l15 * 16 + 8 * hi));
      d0[0] = pack4(sA[0], sA[1], sA[2], sA[3]);
      d0[1] = pack4(sA[4], sA[5], sA[6], sA[7]);
      unsigned long long* d1 =
          (unsigned long long*)(sm + ((16 + l15) * 16 + 8 * hi));
      d1[0] = pack4(sB[0], sB[1], sB[2], sB[3]);
      d1[1] = pack4(sB[4], sB[5], sB[6], sB[7]);
    }
    asm volatile("s_wait_dscnt 0x0" ::: "memory");  // wave-private LDS RAW
    v16bf aP;
#pragma unroll
    for (int i = 0; i < 8; ++i) {
      aP[i]     = __builtin_bit_cast(__bf16, sm[((hi ? 16 : 0) + i) * 16 + l15]);
      aP[i + 8] = __builtin_bit_cast(__bf16, sm[((hi ? 24 : 8) + i) * 16 + l15]);
    }
    // PV: out[16q x 64d] += P[16x32] * V[32x16] per 16-wide d tile (V from LDS)
#pragma unroll
    for (int s4 = 0; s4 < 4; ++s4) {
      v16bf bv = *(const v16bf*)(&ldsV[buf][0] + (s4 * 16 + l15) * 32 + (hi ? 16 : 0));
      o[s4] = wmma_bf16(aP, bv, o[s4]);
    }
  }

  // normalize by row sum (butterfly over the 16 lanes of each half) and store
#pragma unroll
  for (int r = 0; r < 8; ++r) {
    float lr = lrow[r];
#pragma unroll
    for (int off = 1; off < 16; off <<= 1) lr += __shfl_xor(lr, off, 32);
    float inv = 1.0f / lr;
#pragma unroll
    for (int s4 = 0; s4 < 4; ++s4) o[s4][r] *= inv;
  }
#pragma unroll
  for (int s4 = 0; s4 < 4; ++s4)
#pragma unroll
    for (int r = 0; r < 8; ++r)
      out[(size_t)(b * L_ + q0 + r + 8 * hi) * D_ + h * HD_ + s4 * 16 + l15] =
          o[s4][r];
}

// ---------------------------------------------------------------------------
extern "C" void kernel_launch(void* const* d_in, const int* in_sizes, int n_in,
                              void* d_out, int out_size, void* d_ws,
                              size_t ws_size, hipStream_t stream) {
  const float* x    = (const float*)d_in[0];   // [4,1024,1024]
  const float* Wqv  = (const float*)d_in[1];   // [1024,2048]
  const float* rr   = (const float*)d_in[2];   // [16,64] r_r_bias
  const float* rw   = (const float*)d_in[3];   // [16,64] r_w_bias
  const int*   mask = (const int*)d_in[4];     // [4,1024]
  float* out = (float*)d_out;                  // [4,1024,1024]

  // workspace layout (bytes), total ~29.7 MB
  char* w = (char*)d_ws;
  __bf16* pos = (__bf16*)(w);                                    // 2048*64*2
  __bf16* xb  = (__bf16*)(w + 262144);                           // 4096*1024*2
  __bf16* wT  = (__bf16*)(w + 262144 + 8388608);                 // 2048*1024*2
  __bf16* qb  = (__bf16*)(w + 262144 + 8388608 + 4194304);       // 4096*1024*2
  __bf16* vT  = (__bf16*)(w + 262144 + 8388608 + 4194304 + 8388608);

  pos_table_kernel<<<512, 256, 0, stream>>>(pos);
  cvt_x_kernel<<<16384, 256, 0, stream>>>(x, xb);
  cvt_wT_kernel<<<8192, 256, 0, stream>>>(Wqv, wT);
  qv_gemm_kernel<<<dim3(128, 64), 128, 0, stream>>>(xb, wT, qb, vT);
  rel_attn_kernel<<<dim3(64, 16), 128, 0, stream>>>(qb, xb, vT, pos, rr, rw,
                                                    mask, out);
}
